// OrientedRPN_65859028517274
// MI455X (gfx1250) — matrix operand order repaired and checked by reading, not compiled, not run on hardware
//
#include <hip/hip_runtime.h>
#include <hip/hip_bf16.h>
#include <math.h>

typedef __attribute__((ext_vector_type(16))) _Float16 v16h;
typedef __attribute__((ext_vector_type(8)))  float    v8f;

// ---------------- problem constants ----------------
#define NLVL   5
#define T_TOT  8768          // 2000*4 + 768 per batch
#define TOPN   1000
#define OUT_P  0             // (2, 8768, 4, 2)
#define OUT_S  140288        // (2, 8768)
#define OUT_A  157824        // (2, 8768, 4)
#define OUT_F  227968        // (2, 8768)

__host__ __device__ static inline int lvl_S(int l)  { return 256 >> l; }
__host__ __device__ static inline int lvl_K(int l)  { return (l < 4) ? 2000 : 768; }
__host__ __device__ static inline int lvl_off(int l){ return l * 2000; }

// ---- CDNA5 async global->LDS path (guarded: falls back to sync staging) ----
#if defined(__HIP_DEVICE_COMPILE__) && \
    __has_builtin(__builtin_amdgcn_global_load_async_to_lds_b128) && \
    __has_builtin(__builtin_amdgcn_s_wait_asynccnt)
#define ASYNC_G2L 1
#endif

#ifdef ASYNC_G2L
typedef __attribute__((__vector_size__(4 * sizeof(int)))) int i32x4;
__device__ __forceinline__ void g2l_b128(const _Float16* g, _Float16* l)
{
    __builtin_amdgcn_global_load_async_to_lds_b128(
        (__attribute__((address_space(1))) i32x4*)g,
        (__attribute__((address_space(3))) i32x4*)l, 0, 0);
}
#endif

// =====================================================================
// 1) feats fp32 NCHW -> f16 NHWC  (contiguous channels for implicit GEMM)
// =====================================================================
__global__ void cvt_feat_kernel(const float* __restrict__ x,
                                _Float16* __restrict__ xT, int HW)
{
    size_t t = (size_t)blockIdx.x * blockDim.x + threadIdx.x;
    size_t total = (size_t)2 * HW * 256;
    if (t >= total) return;
    int c = (int)(t & 255);
    size_t mp = t >> 8;                  // b*HW + p
    int b = (int)(mp / HW);
    int p = (int)(mp - (size_t)b * HW);
    xT[t] = (_Float16)x[((size_t)b * 256 + c) * HW + p];
}

// =====================================================================
// 2) conv weights (5,256,256,3,3) fp32 -> Wt[l][n][k] f16, k = kpos*256+c
// =====================================================================
__global__ void prep_weights_kernel(const float* __restrict__ convW,
                                    _Float16* __restrict__ Wt)
{
    size_t t = (size_t)blockIdx.x * blockDim.x + threadIdx.x;
    const size_t total = (size_t)NLVL * 256 * 2304;
    if (t >= total) return;
    int l   = (int)(t / (256 * 2304));
    int rem = (int)(t - (size_t)l * 256 * 2304);
    int n = rem / 2304;
    int k = rem - n * 2304;
    int kpos = k >> 8;
    int c    = k & 255;
    Wt[t] = (_Float16)convW[((((size_t)l * 256 + n) * 256) + c) * 9 + kpos];
}

// =====================================================================
// 3) 3x3 SAME conv as implicit GEMM with v_wmma_f32_16x16x32_f16.
//    WG = 256 thr (8 waves), tile M=128, N=128; wave -> 32x64 (8 accums).
//    Double-buffered LDS, async global->LDS staging, 1 barrier / K-step.
// =====================================================================
__global__ __launch_bounds__(256)
void conv3x3_wmma_kernel(const _Float16* __restrict__ xT,  // (B*HW, 256)
                         const _Float16* __restrict__ Wt,  // (256, 2304)
                         const float*    __restrict__ bias,// (256)
                         _Float16*       __restrict__ yT,  // (B*HW, 256)
                         int S, int M)
{
    __shared__ _Float16 Alds[2][128][48];   // 96B rows: 16B-aligned cols
    __shared__ _Float16 Blds[2][128][48];

    const int tid  = threadIdx.x;
    const int lane = tid & 31;
    const int wave = tid >> 5;
    const int wm   = wave >> 1;          // 0..3  (M direction, 32 rows each)
    const int wn   = wave & 1;           // 0..1  (N direction, 64 cols each)
    const int m0   = blockIdx.x * 128;
    const int n0   = blockIdx.y * 128;
    const int HW   = S * S;

    const int srow  = tid >> 1;          // staging row 0..127
    const int shalf = (tid & 1) * 16;    // 16 f16 per thread-half

    v8f zero = {0.f,0.f,0.f,0.f,0.f,0.f,0.f,0.f};
    v8f acc[2][4];
#pragma unroll
    for (int i = 0; i < 2; ++i)
#pragma unroll
        for (int j = 0; j < 4; ++j) acc[i][j] = zero;

    // spatial coords of this thread's A staging row (fixed over K)
    int mg = m0 + srow;
    bool mvalid = (mg < M);
    int b = 0, yy = 0, xx = 0;
    if (mvalid) { b = mg / HW; int rem = mg - b * HW; yy = rem / S; xx = rem - yy * S; }

    const size_t browoff = (size_t)(n0 + srow) * 2304 + shalf;

    // stage one 128x32 A tile + 128x32 B tile for flat K-step s (k = 32*s)
    auto stage = [&](int s, int buf) {
        const int kpos = s >> 3;               // 0..8 -> (dy,dx)
        const int c0   = (s & 7) << 5;         // channel base
        const int kd   = kpos / 3;
        const int dy   = kd - 1;
        const int dx   = kpos - kd * 3 - 1;

        _Float16* adst = &Alds[buf][srow][shalf];
        _Float16* bdst = &Blds[buf][srow][shalf];
        const _Float16* bsrc = Wt + browoff + ((size_t)s << 5);

        bool v = false;
        const _Float16* asrc = nullptr;
        if (mvalid) {
            int iy = yy + dy, ix = xx + dx;
            if (iy >= 0 && iy < S && ix >= 0 && ix < S) {
                v = true;
                asrc = xT + (((size_t)(b * HW + iy * S + ix)) << 8) + c0 + shalf;
            }
        }
#ifdef ASYNC_G2L
        if (v) {
            g2l_b128(asrc, adst);
            g2l_b128(asrc + 8, adst + 8);
        } else {
            uint4 z = make_uint4(0u,0u,0u,0u);
            *(uint4*)adst       = z;
            *(uint4*)(adst + 8) = z;
        }
        g2l_b128(bsrc, bdst);
        g2l_b128(bsrc + 8, bdst + 8);
#else
        uint4 a0 = make_uint4(0u,0u,0u,0u), a1 = a0;
        if (v) {
            a0 = *(const uint4*)asrc;
            a1 = *(const uint4*)(asrc + 8);
            __builtin_prefetch((const void*)(asrc + 32), 0, 0);
        }
        *(uint4*)adst       = a0;
        *(uint4*)(adst + 8) = a1;
        uint4 b0v = *(const uint4*)bsrc;
        uint4 b1v = *(const uint4*)(bsrc + 8);
        *(uint4*)bdst       = b0v;
        *(uint4*)(bdst + 8) = b1v;
#endif
    };

    const int rsel     = (lane < 16) ? 0 : 1;
    const int frag_row = lane & 15;

    stage(0, 0);                          // prologue fill of buffer 0

    for (int s = 0; s < 72; ++s) {
        const int buf = s & 1;
#ifdef ASYNC_G2L
        __builtin_amdgcn_s_wait_asynccnt(0);   // own loads into buf complete
#endif
        __syncthreads();                        // publish buf; retire reads of buf^1
        if (s + 1 < 72) stage(s + 1, buf ^ 1);  // prefetch next tile (overlaps WMMA)

        // ---- fragments (ISA 7.12.2 wave32 16-bit layouts) ----
        v16h afrag[2];
#pragma unroll
        for (int ms = 0; ms < 2; ++ms) {
            const _Float16* ap = &Alds[buf][wm * 32 + ms * 16 + frag_row][rsel ? 8 : 0];
            uint4* d = (uint4*)&afrag[ms];
            d[0] = *(const uint4*)ap;            // K base..base+7
            d[1] = *(const uint4*)(ap + 16);     // K base+16..base+23
        }
        v16h bfrag[4];
#pragma unroll
        for (int ns = 0; ns < 4; ++ns) {
            const _Float16* bp = &Blds[buf][wn * 64 + ns * 16 + frag_row][rsel ? 8 : 0];
            uint4* d = (uint4*)&bfrag[ns];
            d[0] = *(const uint4*)bp;            // K base..base+7
            d[1] = *(const uint4*)(bp + 16);     // K base+16..base+23
        }

#pragma unroll
        for (int ms = 0; ms < 2; ++ms)
#pragma unroll
            for (int ns = 0; ns < 4; ++ns)
                acc[ms][ns] = __builtin_amdgcn_wmma_f32_16x16x32_f16(
                    false, afrag[ms], false, bfrag[ns],
                    (short)0, acc[ms][ns], false, false);
    }

    // ---- epilogue: +bias, f32->f16, NHWC store ----
#pragma unroll
    for (int ms = 0; ms < 2; ++ms) {
#pragma unroll
        for (int ns = 0; ns < 4; ++ns) {
            int ng = n0 + wn * 64 + ns * 16 + (lane & 15);
            float bv = bias[ng];
#pragma unroll
            for (int r = 0; r < 8; ++r) {
                int ml = wm * 32 + ms * 16 + r + ((lane < 16) ? 0 : 8);
                int mo = m0 + ml;
                if (mo < M)
                    yT[(size_t)mo * 256 + ng] = (_Float16)(acc[ms][ns][r] + bv);
            }
        }
    }
}

// =====================================================================
// 4) 1x1 heads: 18 reg + 3 obj channels; weights live in LDS
// =====================================================================
__global__ __launch_bounds__(256)
void head_kernel(const _Float16* __restrict__ yT,
                 const float* __restrict__ regW, const float* __restrict__ regB,
                 const float* __restrict__ objW, const float* __restrict__ objB,
                 float* __restrict__ off,   // (2, 3*HW, 6)
                 float* __restrict__ obj,   // (2, 3*HW)
                 int S)
{
    __shared__ float Wl[21][256];
    int tid = threadIdx.x;
    for (int i = tid; i < 18 * 256; i += 256) Wl[i >> 8][i & 255] = regW[i];
    for (int i = tid; i < 3 * 256;  i += 256) Wl[18 + (i >> 8)][i & 255] = objW[i];
    __syncthreads();

    int HW = S * S, M = 2 * HW;
    int m = blockIdx.x * 256 + tid;
    if (m >= M) return;

    const _Float16* y = yT + (size_t)m * 256;
    float s[21];
#pragma unroll
    for (int j = 0; j < 21; ++j) s[j] = 0.f;
    for (int c = 0; c < 256; ++c) {
        float v = (float)y[c];
#pragma unroll
        for (int j = 0; j < 21; ++j) s[j] += v * Wl[j][c];
    }
    int b = m / HW, p = m - b * HW;
    const float std6[6] = {1.f, 1.f, 1.f, 1.f, 0.5f, 0.5f};
    for (int a = 0; a < 3; ++a) {
        size_t n = (size_t)b * (3 * HW) + (size_t)a * HW + p;
        for (int d = 0; d < 6; ++d)
            off[n * 6 + d] = (s[a * 6 + d] + regB[a * 6 + d]) * std6[d];
        obj[n] = s[18 + a] + objB[a];
    }
}

// =====================================================================
// 5) midpoint-offset decode -> vertices + HBB
// =====================================================================
__global__ void decode_kernel(const float* __restrict__ off,
                              float* __restrict__ verts,  // (2, NA, 8)
                              float* __restrict__ hbb,    // (2, NA, 4)
                              int S)
{
    int HW = S * S, NA = 3 * HW;
    int t = blockIdx.x * blockDim.x + threadIdx.x;
    if (t >= 2 * NA) return;
    int n = t % NA;
    int a = n / HW, rem = n - a * HW, yy = rem / S, xx = rem - yy * S;
    float ratio = (a == 0) ? 0.5f : ((a == 1) ? 1.0f : 2.0f);
    float sr = sqrtf(ratio);
    float aw = 8.0f / sr, ah = 8.0f * sr;
    float cx0 = xx + 0.5f, cy0 = yy + 0.5f;
    const float* o = off + (size_t)t * 6;
    float dx = o[0], dy = o[1], dw = o[2], dh = o[3], da = o[4], db = o[5];
    float w  = aw * expf(dw), h = ah * expf(dh);
    float cx = cx0 + dx * aw, cy = cy0 + dy * ah;
    float wa = da * w, hb = db * h;
    float v[8] = { cx + wa,        cy - h * 0.5f,
                   cx + w * 0.5f,  cy + hb,
                   cx - wa,        cy + h * 0.5f,
                   cx - w * 0.5f,  cy - hb };
    float* vo = verts + (size_t)t * 8;
#pragma unroll
    for (int k = 0; k < 8; ++k) vo[k] = v[k];
    float* ho = hbb + (size_t)t * 4;
    ho[0] = fminf(fminf(v[0], v[2]), fminf(v[4], v[6]));
    ho[1] = fminf(fminf(v[1], v[3]), fminf(v[5], v[7]));
    ho[2] = fmaxf(fmaxf(v[0], v[2]), fmaxf(v[4], v[6]));
    ho[3] = fmaxf(fmaxf(v[1], v[3]), fmaxf(v[5], v[7]));
}

// =====================================================================
// 6) per-(batch,level) top-K: histogram threshold + compaction + bitonic
// =====================================================================
#define CANDK 4096
__global__ __launch_bounds__(1024)
void topk_kernel(const float* __restrict__ obj,   // (2, N)
                 float* __restrict__ sc_all,      // (2, T_TOT)
                 unsigned* __restrict__ idx_all,  // (2, T_TOT)
                 int N, int K, int loff)
{
    __shared__ unsigned hist[2048];
    __shared__ unsigned ckey[CANDK];
    __shared__ unsigned cidx[CANDK];
    __shared__ unsigned s_thr, s_cnt;
    int b = blockIdx.x, tid = threadIdx.x, nt = blockDim.x;
    const float* sc = obj + (size_t)b * N;

    for (int i = tid; i < 2048; i += nt) hist[i] = 0u;
    if (tid == 0) s_cnt = 0u;
    __syncthreads();
    for (int i = tid; i < N; i += nt) {
        unsigned u = __float_as_uint(sc[i]);
        u = (u & 0x80000000u) ? ~u : (u | 0x80000000u);
        atomicAdd(&hist[u >> 21], 1u);
    }
    __syncthreads();
    if (tid == 0) {
        unsigned cum = 0; int bin = 0;
        for (int i = 2047; i >= 0; --i) { cum += hist[i]; if (cum >= (unsigned)K) { bin = i; break; } }
        s_thr = (unsigned)bin;
    }
    __syncthreads();
    unsigned thr = s_thr;
    for (int i = tid; i < N; i += nt) {
        unsigned u = __float_as_uint(sc[i]);
        u = (u & 0x80000000u) ? ~u : (u | 0x80000000u);
        if ((u >> 21) >= thr) {
            unsigned p = atomicAdd(&s_cnt, 1u);
            if (p < CANDK) { ckey[p] = u; cidx[p] = (unsigned)i; }
        }
    }
    __syncthreads();
    unsigned nc = s_cnt; if (nc > CANDK) nc = CANDK;
    for (int i = tid; i < CANDK; i += nt) if (i >= (int)nc) { ckey[i] = 0u; cidx[i] = 0u; }
    __syncthreads();
    // bitonic sort, descending
    for (int k = 2; k <= CANDK; k <<= 1) {
        for (int j = k >> 1; j > 0; j >>= 1) {
            for (int i = tid; i < CANDK; i += nt) {
                int ixj = i ^ j;
                if (ixj > i) {
                    bool descBlk = ((i & k) == 0);
                    unsigned a = ckey[i], c = ckey[ixj];
                    bool sw = descBlk ? (a < c) : (a > c);
                    if (sw) {
                        ckey[i] = c; ckey[ixj] = a;
                        unsigned t2 = cidx[i]; cidx[i] = cidx[ixj]; cidx[ixj] = t2;
                    }
                }
            }
            __syncthreads();
        }
    }
    for (int i = tid; i < K; i += nt) {
        unsigned u = ckey[i];
        unsigned bits = (u & 0x80000000u) ? (u & 0x7fffffffu) : ~u;
        sc_all[(size_t)b * T_TOT + loff + i]  = __uint_as_float(bits);
        idx_all[(size_t)b * T_TOT + loff + i] = cidx[i];
    }
}

// =====================================================================
// 7) sequential-suppression NMS on sorted HBBs (per batch,level)
// =====================================================================
__global__ __launch_bounds__(1024)
void nms_kernel(const float* __restrict__ hbb,        // (2, NA, 4)
                const unsigned* __restrict__ idx_all, // (2, T_TOT)
                unsigned* __restrict__ keep_all,      // (2, T_TOT)
                int NA, int K, int loff)
{
    __shared__ float bx[2000][4];
    __shared__ float area[2000];
    __shared__ unsigned char keep[2048];
    int b = blockIdx.x, tid = threadIdx.x, nt = blockDim.x;
    for (int i = tid; i < K; i += nt) {
        unsigned n = idx_all[(size_t)b * T_TOT + loff + i];
        const float* p = hbb + ((size_t)b * NA + n) * 4;
        float x1 = p[0], y1 = p[1], x2 = p[2], y2 = p[3];
        bx[i][0] = x1; bx[i][1] = y1; bx[i][2] = x2; bx[i][3] = y2;
        area[i] = (x2 - x1) * (y2 - y1);
        keep[i] = 1;
    }
    __syncthreads();
    for (int i = 0; i < K - 1; ++i) {
        if (keep[i]) {
            float x1 = bx[i][0], y1 = bx[i][1], x2 = bx[i][2], y2 = bx[i][3], ai = area[i];
            for (int j = i + 1 + tid; j < K; j += nt) {
                if (keep[j]) {
                    float lx = fmaxf(x1, bx[j][0]);
                    float ly = fmaxf(y1, bx[j][1]);
                    float rx = fminf(x2, bx[j][2]);
                    float ry = fminf(y2, bx[j][3]);
                    float w = fmaxf(rx - lx, 0.f), h = fmaxf(ry - ly, 0.f);
                    float inter = w * h;
                    float iou = inter / (ai + area[j] - inter + 1e-8f);
                    if (iou > 0.5f) keep[j] = 0;
                }
            }
        }
        __syncthreads();
    }
    for (int i = tid; i < K; i += nt)
        keep_all[(size_t)b * T_TOT + loff + i] = keep[i];
}

// =====================================================================
// 8) cross-level top-1000 of NMS-masked scores -> final mask
// =====================================================================
__global__ __launch_bounds__(1024)
void final_select_kernel(const float* __restrict__ sc_all,
                         const unsigned* __restrict__ keep_all,
                         unsigned* __restrict__ final_all)
{
    const int C2 = 2048;
    __shared__ unsigned hist[2048];
    __shared__ unsigned ckey[2048];
    __shared__ unsigned cidx[2048];
    __shared__ unsigned s_thr, s_cnt;
    int b = blockIdx.x, tid = threadIdx.x, nt = blockDim.x;
    for (int i = tid; i < T_TOT; i += nt) final_all[(size_t)b * T_TOT + i] = 0u;
    for (int i = tid; i < 2048; i += nt) hist[i] = 0u;
    if (tid == 0) s_cnt = 0u;
    __syncthreads();
    for (int i = tid; i < T_TOT; i += nt) {
        if (!keep_all[(size_t)b * T_TOT + i]) continue;
        unsigned u = __float_as_uint(sc_all[(size_t)b * T_TOT + i]);
        u = (u & 0x80000000u) ? ~u : (u | 0x80000000u);
        if (!u) u = 1u;
        atomicAdd(&hist[u >> 21], 1u);
    }
    __syncthreads();
    if (tid == 0) {
        unsigned cum = 0; int bin = 0;
        for (int i = 2047; i >= 0; --i) { cum += hist[i]; if (cum >= (unsigned)TOPN) { bin = i; break; } }
        s_thr = (unsigned)bin;
    }
    __syncthreads();
    unsigned thr = s_thr;
    for (int i = tid; i < T_TOT; i += nt) {
        if (!keep_all[(size_t)b * T_TOT + i]) continue;
        unsigned u = __float_as_uint(sc_all[(size_t)b * T_TOT + i]);
        u = (u & 0x80000000u) ? ~u : (u | 0x80000000u);
        if (!u) u = 1u;
        if ((u >> 21) >= thr) {
            unsigned p = atomicAdd(&s_cnt, 1u);
            if (p < C2) { ckey[p] = u; cidx[p] = (unsigned)i; }
        }
    }
    __syncthreads();
    unsigned nc = s_cnt; if (nc > (unsigned)C2) nc = C2;
    for (int i = tid; i < C2; i += nt) if (i >= (int)nc) { ckey[i] = 0u; cidx[i] = 0u; }
    __syncthreads();
    for (int k = 2; k <= C2; k <<= 1) {
        for (int j = k >> 1; j > 0; j >>= 1) {
            for (int i = tid; i < C2; i += nt) {
                int ixj = i ^ j;
                if (ixj > i) {
                    bool descBlk = ((i & k) == 0);
                    unsigned a = ckey[i], c = ckey[ixj];
                    bool sw = descBlk ? (a < c) : (a > c);
                    if (sw) {
                        ckey[i] = c; ckey[ixj] = a;
                        unsigned t2 = cidx[i]; cidx[i] = cidx[ixj]; cidx[ixj] = t2;
                    }
                }
            }
            __syncthreads();
        }
    }
    for (int i = tid; i < TOPN; i += nt)
        if (i < (int)nc && ckey[i] != 0u)
            final_all[(size_t)b * T_TOT + cidx[i]] = 1u;
}

// =====================================================================
// 9) gather and write P / S / A / final flat into d_out
// =====================================================================
__global__ void write_output_kernel(const float* __restrict__ v0, const float* __restrict__ v1,
                                    const float* __restrict__ v2, const float* __restrict__ v3,
                                    const float* __restrict__ v4,
                                    const float* __restrict__ sc_all,
                                    const unsigned* __restrict__ idx_all,
                                    const unsigned* __restrict__ final_all,
                                    float* __restrict__ out)
{
    int t = blockIdx.x * blockDim.x + threadIdx.x;
    if (t >= 2 * T_TOT) return;
    int b = t / T_TOT, e = t - b * T_TOT;
    int l = e / 2000; if (l > 4) l = 4;
    int S = lvl_S(l), HW = S * S, NA = 3 * HW;
    const float* vl = (l == 0) ? v0 : (l == 1) ? v1 : (l == 2) ? v2 : (l == 3) ? v3 : v4;
    unsigned n = idx_all[t];
    const float* vp = vl + ((size_t)b * NA + n) * 8;
    float* po = out + OUT_P + (size_t)t * 8;
#pragma unroll
    for (int k = 0; k < 8; ++k) po[k] = vp[k];
    out[OUT_S + t] = sc_all[t];
    int a = n / HW, rem = n - a * HW, yy = rem / S, xx = rem - yy * S;
    float ratio = (a == 0) ? 0.5f : ((a == 1) ? 1.0f : 2.0f);
    float sr = sqrtf(ratio);
    float* ao = out + OUT_A + (size_t)t * 4;
    ao[0] = xx + 0.5f; ao[1] = yy + 0.5f; ao[2] = 8.0f / sr; ao[3] = 8.0f * sr;
    out[OUT_F + t] = final_all[t] ? 1.0f : 0.0f;
}

// =====================================================================
// host-side launcher
// =====================================================================
extern "C" void kernel_launch(void* const* d_in, const int* in_sizes, int n_in,
                              void* d_out, int out_size, void* d_ws, size_t ws_size,
                              hipStream_t stream)
{
    const float* feat[NLVL];
    for (int l = 0; l < NLVL; ++l) feat[l] = (const float*)d_in[l];
    const float* convW = (const float*)d_in[5];
    const float* convB = (const float*)d_in[6];
    const float* regW  = (const float*)d_in[7];
    const float* regB  = (const float*)d_in[8];
    const float* objW  = (const float*)d_in[9];
    const float* objB  = (const float*)d_in[10];
    float* out = (float*)d_out;

    // ---- carve workspace ----
    char* ws = (char*)d_ws;
    size_t cur = 0;
    auto carve = [&](size_t bytes) -> char* {
        char* p = ws + cur;
        cur = (cur + bytes + 255) & ~(size_t)255;
        return p;
    };
    _Float16* featT[NLVL];  _Float16* yT[NLVL];
    float *offp[NLVL], *objp[NLVL], *vertsp[NLVL], *hbbp[NLVL];
    for (int l = 0; l < NLVL; ++l) {
        int HW = lvl_S(l) * lvl_S(l);
        featT[l] = (_Float16*)carve((size_t)2 * HW * 256 * sizeof(_Float16));
        yT[l]    = (_Float16*)carve((size_t)2 * HW * 256 * sizeof(_Float16));
        offp[l]  = (float*)carve((size_t)2 * 3 * HW * 6 * sizeof(float));
        objp[l]  = (float*)carve((size_t)2 * 3 * HW * sizeof(float));
        vertsp[l]= (float*)carve((size_t)2 * 3 * HW * 8 * sizeof(float));
        hbbp[l]  = (float*)carve((size_t)2 * 3 * HW * 4 * sizeof(float));
    }
    _Float16* Wt = (_Float16*)carve((size_t)NLVL * 256 * 2304 * sizeof(_Float16));
    float*    sc_all    = (float*)carve((size_t)2 * T_TOT * sizeof(float));
    unsigned* idx_all   = (unsigned*)carve((size_t)2 * T_TOT * sizeof(unsigned));
    unsigned* keep_all  = (unsigned*)carve((size_t)2 * T_TOT * sizeof(unsigned));
    unsigned* final_all = (unsigned*)carve((size_t)2 * T_TOT * sizeof(unsigned));
    if (cur > ws_size) return;   // workspace too small: bail deterministically

    // ---- 1. weight prep (once) ----
    {
        size_t total = (size_t)NLVL * 256 * 2304;
        prep_weights_kernel<<<(unsigned)((total + 255) / 256), 256, 0, stream>>>(convW, Wt);
    }

    // ---- 2..5 per level: convert, conv(WMMA), heads, decode ----
    for (int l = 0; l < NLVL; ++l) {
        int S = lvl_S(l), HW = S * S, M = 2 * HW;
        {
            size_t total = (size_t)2 * HW * 256;
            cvt_feat_kernel<<<(unsigned)((total + 255) / 256), 256, 0, stream>>>(feat[l], featT[l], HW);
        }
        {
            dim3 grid((M + 127) / 128, 2, 1);
            conv3x3_wmma_kernel<<<grid, 256, 0, stream>>>(
                featT[l], Wt + (size_t)l * 256 * 2304, convB + l * 256, yT[l], S, M);
        }
        head_kernel<<<(M + 255) / 256, 256, 0, stream>>>(
            yT[l], regW + (size_t)l * 18 * 256, regB + l * 18,
            objW + (size_t)l * 3 * 256, objB + l * 3, offp[l], objp[l], S);
        {
            int total = 2 * 3 * HW;
            decode_kernel<<<(total + 255) / 256, 256, 0, stream>>>(offp[l], vertsp[l], hbbp[l], S);
        }
    }

    // ---- 6..7 per level: top-K then NMS ----
    for (int l = 0; l < NLVL; ++l) {
        int S = lvl_S(l), NA = 3 * S * S, K = lvl_K(l), lo = lvl_off(l);
        topk_kernel<<<2, 1024, 0, stream>>>(objp[l], sc_all, idx_all, NA, K, lo);
        nms_kernel<<<2, 1024, 0, stream>>>(hbbp[l], idx_all, keep_all, NA, K, lo);
    }

    // ---- 8. cross-level top-1000 ----
    final_select_kernel<<<2, 1024, 0, stream>>>(sc_all, keep_all, final_all);

    // ---- 9. emit outputs ----
    write_output_kernel<<<(2 * T_TOT + 255) / 256, 256, 0, stream>>>(
        vertsp[0], vertsp[1], vertsp[2], vertsp[3], vertsp[4],
        sc_all, idx_all, final_all, out);
}
